// GCN_25331717111859
// MI455X (gfx1250) — matrix operand order repaired and checked by reading, not compiled, not run on hardware
//
#include <hip/hip_runtime.h>
#include <math.h>

// ---------------------------------------------------------------- types ----
typedef __bf16  bf16_t;
typedef __bf16  v16bf __attribute__((ext_vector_type(16)));
typedef float   v8f   __attribute__((ext_vector_type(8)));

#define WMMA_BF16(A, B, C) \
    __builtin_amdgcn_wmma_f32_16x16x32_bf16(false, (A), false, (B), (short)0, (C), false, false)

// ------------------------------------------------------------- constants ---
static constexpr int N_NODES  = 20000;
static constexpr int N_EDGES  = 320000;
static constexpr int F_IN     = 64;
static constexpr int C        = 64;
static constexpr int H        = 4;
static constexpr int HC       = 256;       // H*C
static constexpr int L        = 3;
static constexpr int N_RES    = 2;
static constexpr int N_PAIRS  = 8192;
static constexpr int EDGE_DIM = 9;
static constexpr int HID      = 4096;      // 64*C
static constexpr int NEXP     = 8;
static constexpr int TWO_C    = 128;
static constexpr int MAX_TILES = N_PAIRS / 16 + NEXP;     // 520
static constexpr int PAD_ROWS  = N_PAIRS + NEXP * 16;     // 8320

// meta int-array layout
static constexpr int META_NT   = 0;
static constexpr int META_TE   = 1;                 // tile_expert[520]
static constexpr int META_TR   = 1 + MAX_TILES;     // tile_rowoff[520]
static constexpr int META_BASE = META_TR + MAX_TILES;   // base[9]
static constexpr int META_PAD  = META_BASE + 9;         // padbase[9]
static constexpr int META_CNT  = META_PAD + 9;          // count[8]

// workspace offsets (bytes)
static constexpr size_t OFF_H    = 0;
static constexpr size_t OFF_H2   = OFF_H   + (size_t)N_NODES * F_IN * 4;       // 5.12MB
static constexpr size_t OFF_XL   = OFF_H2  + (size_t)N_NODES * F_IN * 4;
static constexpr size_t OFF_XR   = OFF_XL  + (size_t)N_NODES * HC * 4;         // 20.48MB
static constexpr size_t OFF_AGG  = OFF_XR  + (size_t)N_NODES * HC * 4;
static constexpr size_t OFF_LOG  = OFF_AGG + (size_t)N_NODES * HC * 4;
static constexpr size_t OFF_MX   = OFF_LOG + (size_t)N_EDGES * H * 4;
static constexpr size_t OFF_DEN  = OFF_MX  + (size_t)N_NODES * H * 4;
static constexpr size_t OFF_ZS   = OFF_DEN + (size_t)N_NODES * H * 4;
static constexpr size_t OFF_RANK = OFF_ZS  + (size_t)PAD_ROWS * TWO_C * 4;
static constexpr size_t OFF_META = OFF_RANK + (size_t)N_PAIRS * 4;
static constexpr size_t OFF_WLP  = OFF_META + 8192;
static constexpr size_t OFF_WRP  = OFF_WLP + (size_t)L * F_IN * HC * 2;
static constexpr size_t OFF_W1P  = OFF_WRP + (size_t)L * F_IN * HC * 2;
static constexpr size_t OFF_W2P  = OFF_W1P + (size_t)NEXP * N_RES * TWO_C * HID * 2;
// total ~116 MB

// ------------------------------------------------------------- helpers -----
__device__ __forceinline__ unsigned f32_ord(float f) {
    unsigned u = __float_as_uint(f);
    return (u & 0x80000000u) ? ~u : (u | 0x80000000u);
}
__device__ __forceinline__ float ord_f32(unsigned u) {
    return (u & 0x80000000u) ? __uint_as_float(u & 0x7fffffffu) : __uint_as_float(~u);
}
__device__ __forceinline__ float sigmoidf_d(float x) { return 1.f / (1.f + __expf(-x)); }

// A-matrix fragment from LDS (16-bit A 16x32 layout, CDNA5 ISA 7.12.2)
__device__ __forceinline__ v16bf load_a_frag(const bf16_t* base, int ld, int kbase, int lane) {
    int m  = lane & 15;
    int kh = (lane >> 4) << 3;            // 0 or 8
    const bf16_t* row = base + (size_t)m * ld;
    v16bf a;
#pragma unroll
    for (int i = 0; i < 8; ++i) a[i]     = row[kbase + kh + i];
#pragma unroll
    for (int i = 0; i < 8; ++i) a[8 + i] = row[kbase + 16 + kh + i];
    return a;
}

// B-matrix fragment from packed (fragment-major) weights: 32B contiguous per lane
__device__ __forceinline__ v16bf load_b_frag(const bf16_t* packed, int ntile, int ktile,
                                             int ktiles, int lane) {
    const bf16_t* p = packed + (((size_t)(ntile * ktiles + ktile) * 32) + lane) * 16;
    return *(const v16bf*)p;
}

// ---------------------------------------------------------------- fills ----
__global__ void k_fill_f32(float* p, float v, int n) {
    int i = blockIdx.x * 256 + threadIdx.x;
    if (i < n) p[i] = v;
}
__global__ void k_fill_u32(unsigned* p, unsigned v, int n) {
    int i = blockIdx.x * 256 + threadIdx.x;
    if (i < n) p[i] = v;
}

// ------------------------------------------------------- weight packing ----
// src: row-major [K x N] f32; dst: fragment-major bf16 (per B layout above).
__global__ void k_pack_batched(const float* __restrict__ src, bf16_t* __restrict__ dst,
                               int K, int N) {
    size_t mat = blockIdx.y;
    src += mat * (size_t)K * N;
    dst += mat * (size_t)K * N;
    int tid    = blockIdx.x * 256 + threadIdx.x;
    int ktiles = K >> 5;
    int lane   = tid & 31;
    int ft     = tid >> 5;                       // ntile*ktiles + ktile
    if (ft >= (N >> 4) * ktiles) return;
    int ktile = ft % ktiles, ntile = ft / ktiles;
    int n  = ntile * 16 + (lane & 15);
    int kb = ktile * 32 + ((lane >> 4) << 3);
    bf16_t* p = dst + ((size_t)ft * 32 + lane) * 16;
#pragma unroll
    for (int i = 0; i < 16; ++i) p[i] = (bf16_t)src[(size_t)(kb + (i < 8 ? i : 8 + i)) * N + n];
}

// ------------------------------------------------------ node transform -----
// out[16 x 256] = in[16 x 64] @ W[64 x 256] + bias      (bf16 WMMA, f32 accum)
__global__ __launch_bounds__(128) void k_node_gemm(const float* __restrict__ in,
                                                   const bf16_t* __restrict__ wpack,
                                                   const float* __restrict__ bias,
                                                   float* __restrict__ out) {
    __shared__ bf16_t Ash[16][72];
    int tid = threadIdx.x;
    int rowbase = blockIdx.x * 16;
    for (int idx = tid; idx < 16 * 64; idx += 128) {
        int r = idx >> 6, c = idx & 63;
        Ash[r][c] = (bf16_t)in[(size_t)(rowbase + r) * 64 + c];
    }
    __syncthreads();
    int wave = tid >> 5, lane = tid & 31;
    v16bf a0 = load_a_frag(&Ash[0][0], 72, 0,  lane);
    v16bf a1 = load_a_frag(&Ash[0][0], 72, 32, lane);
#pragma unroll
    for (int t = 0; t < 4; ++t) {
        int ct = wave * 4 + t;
        v8f acc = {};
        v16bf b0 = load_b_frag(wpack, ct, 0, 2, lane);
        v16bf b1 = load_b_frag(wpack, ct, 1, 2, lane);
        acc = WMMA_BF16(a0, b0, acc);
        acc = WMMA_BF16(a1, b1, acc);
        int col = ct * 16 + (lane & 15);
        float bv = bias[col];
        int rb = rowbase + ((lane >> 4) << 3);
#pragma unroll
        for (int k = 0; k < 8; ++k)
            out[(size_t)(rb + k) * HC + col] = acc[k] + bv;
    }
}

// ----------------------------------------------------------- edge passes ---
__global__ void k_edge_logits(const int* __restrict__ ei, const float* __restrict__ ea,
                              const float* __restrict__ xl, const float* __restrict__ xr,
                              const float* __restrict__ We, const float* __restrict__ att,
                              float* __restrict__ logits, unsigned* __restrict__ mxenc) {
    __shared__ float WeSh[EDGE_DIM][HC];
    __shared__ float attSh[H][C];
    int tid = threadIdx.x;
    for (int idx = tid; idx < EDGE_DIM * HC; idx += 256) WeSh[idx / HC][idx % HC] = We[idx];
    for (int idx = tid; idx < HC; idx += 256) attSh[idx >> 6][idx & 63] = att[idx];
    __syncthreads();
    int g = blockIdx.x * 256 + tid;          // (edge, head), E*H exact
    int e = g >> 2, hh = g & 3;
    int src = ei[e], tgt = ei[N_EDGES + e];
    float eav[EDGE_DIM];
#pragma unroll
    for (int d = 0; d < EDGE_DIM; ++d) eav[d] = ea[(size_t)e * EDGE_DIM + d];
    const float* xlp = xl + (size_t)src * HC + hh * 64;
    const float* xrp = xr + (size_t)tgt * HC + hh * 64;
    float logit = 0.f;
    for (int c = 0; c < 64; ++c) {
        float ev = 0.f;
#pragma unroll
        for (int d = 0; d < EDGE_DIM; ++d) ev += eav[d] * WeSh[d][hh * 64 + c];
        float m = xlp[c] + xrp[c] + ev;
        m = m > 0.f ? m : 0.2f * m;          // leaky_relu(0.2)
        logit += m * attSh[hh][c];
    }
    logits[(size_t)e * H + hh] = logit;
    atomicMax(&mxenc[(size_t)tgt * H + hh], f32_ord(logit));
}

__global__ void k_edge_softmax(const int* __restrict__ ei, float* __restrict__ logits,
                               const unsigned* __restrict__ mxenc, float* __restrict__ den) {
    int g = blockIdx.x * 256 + threadIdx.x;
    int e = g >> 2, hh = g & 3;
    int tgt = ei[N_EDGES + e];
    float mx = ord_f32(mxenc[(size_t)tgt * H + hh]);
    float a = __expf(logits[(size_t)e * H + hh] - mx);
    logits[(size_t)e * H + hh] = a;          // in-place -> unnormalized alpha
    atomicAdd(&den[(size_t)tgt * H + hh], a);
}

__global__ void k_edge_agg(const int* __restrict__ ei, const float* __restrict__ alpha,
                           const float* __restrict__ den, const float* __restrict__ xl,
                           float* __restrict__ agg) {
    int g = blockIdx.x * 256 + threadIdx.x;
    int e = g >> 2, hh = g & 3;
    int src = ei[e], tgt = ei[N_EDGES + e];
    float al = alpha[(size_t)e * H + hh] / den[(size_t)tgt * H + hh];
    const float* xlp = xl + (size_t)src * HC + hh * 64;
    float* ap = agg + (size_t)tgt * HC + hh * 64;
    for (int c = 0; c < 64; ++c) atomicAdd(&ap[c], xlp[c] * al);
}

__global__ void k_finalize(const float* __restrict__ agg, const float* __restrict__ bias,
                           const float* __restrict__ gates, int layer,
                           const float* __restrict__ h2, float* __restrict__ h) {
    int idx = blockIdx.x * 256 + threadIdx.x;   // N*64 exact
    int n = idx >> 6, c = idx & 63;
    const float* ap = agg + (size_t)n * HC;
    float v = 0.25f * (ap[c] + ap[64 + c] + ap[128 + c] + ap[192 + c]) + bias[c];
    if (layer == 0) {
        h[idx] = v;
    } else {
        float g = sigmoidf_d(gates[layer - 1]);
        h[idx] = g * v + (1.f - g) * h2[idx];
    }
}

__global__ void k_lrelu(const float* __restrict__ in, float* __restrict__ out, int n) {
    int i = blockIdx.x * 256 + threadIdx.x;
    if (i >= n) return;
    float v = in[i];
    out[i] = v > 0.f ? v : 0.01f * v;
}

// --------------------------------------------------------- pair routing ----
__global__ void k_rank(const int* __restrict__ types, int* __restrict__ rank,
                       int* __restrict__ count) {
    int p = blockIdx.x * 256 + threadIdx.x;
    if (p >= N_PAIRS) return;
    int t = types[p], r = 0;
    for (int q = 0; q < p; ++q) r += (types[q] == t);
    rank[p] = r;
    atomicAdd(&count[t], 1);
}

__global__ void k_meta(int* __restrict__ M) {
    if (threadIdx.x | blockIdx.x) return;
    int base = 0, pbase = 0, nt = 0;
    M[META_BASE] = 0;
    M[META_PAD]  = 0;
    for (int e = 0; e < NEXP; ++e) {
        int c = M[META_CNT + e];
        int t = (c + 15) >> 4;
        for (int k = 0; k < t; ++k) { M[META_TE + nt] = e; M[META_TR + nt] = k * 16; ++nt; }
        base  += c;
        pbase += t * 16;
        M[META_BASE + e + 1] = base;
        M[META_PAD + e + 1]  = pbase;
    }
    M[META_NT] = nt;
}

__global__ void k_gather(const int* __restrict__ edge_cross, const int* __restrict__ types,
                         const int* __restrict__ rank, const int* __restrict__ M,
                         const float* __restrict__ h, float* __restrict__ zs) {
    int g = blockIdx.x * 256 + threadIdx.x;   // P*128 exact
    int p = g >> 7, j = g & 127;
    int e = types[p];
    int pos = M[META_PAD + e] + rank[p];
    int node = edge_cross[p * 2 + (j & 1)];
    zs[(size_t)pos * TWO_C + j] = h[(size_t)node * F_IN + (j >> 1)];
}

// ----------------------------------------------------------- expert MLP ----
__global__ __launch_bounds__(256) void k_expert(const float* __restrict__ zs,
                                                const bf16_t* __restrict__ w1p,
                                                const bf16_t* __restrict__ w2p,
                                                const float* __restrict__ b1,
                                                const float* __restrict__ b2,
                                                const float* __restrict__ gates,
                                                const float* __restrict__ headW,
                                                const float* __restrict__ headb,
                                                const int* __restrict__ M,
                                                float* __restrict__ out) {
    int t = blockIdx.x;
    if (t >= M[META_NT]) return;
    int e      = M[META_TE + t];
    int rowoff = M[META_TR + t];
    int prow0  = M[META_PAD + e] + rowoff;

    __shared__ float  zm[16][TWO_C];     // f32 master tile
    __shared__ bf16_t zr[16][136];       // bf16 relu'd A source
    __shared__ bf16_t tb[16][1040];      // hidden-chunk activations (A for GEMM2)
    __shared__ float  red[16][16];

    int tid = threadIdx.x, wave = tid >> 5, lane = tid & 31;

    for (int idx = tid; idx < 16 * TWO_C; idx += 256)
        zm[idx >> 7][idx & 127] = zs[(size_t)(prow0 + (idx >> 7)) * TWO_C + (idx & 127)];
    __syncthreads();

    for (int r = 0; r < N_RES; ++r) {
        // z = relu(z); bf16 copy for WMMA
        for (int idx = tid; idx < 16 * TWO_C; idx += 256) {
            int rr = idx >> 7, cc = idx & 127;
            float v = zm[rr][cc];
            v = v > 0.f ? v : 0.f;
            zm[rr][cc] = v;
            zr[rr][cc] = (bf16_t)v;
        }
        __syncthreads();

        const bf16_t* W1 = w1p + (size_t)(e * N_RES + r) * TWO_C * HID;
        const bf16_t* W2 = w2p + (size_t)(e * N_RES + r) * HID * TWO_C;
        const float*  B1 = b1  + (size_t)(e * N_RES + r) * HID;
        const float*  B2 = b2  + (size_t)(e * N_RES + r) * TWO_C;
        float gg = sigmoidf_d(gates[e * N_RES + r]);

        v16bf az[4];
#pragma unroll
        for (int ks = 0; ks < 4; ++ks) az[ks] = load_a_frag(&zr[0][0], 136, ks * 32, lane);

        v8f oacc = {};
        for (int chunk = 0; chunk < 4; ++chunk) {
            // phase 1: y-chunk = relu(zr @ W1[:, chunk] + b1) -> tb (bf16)
#pragma unroll
            for (int ct = 0; ct < 8; ++ct) {
                int ntile = chunk * 64 + wave * 8 + ct;     // of 256 N-tiles
                v8f acc = {};
#pragma unroll
                for (int ks = 0; ks < 4; ++ks) {
                    v16bf b = load_b_frag(W1, ntile, ks, 4, lane);
                    acc = WMMA_BF16(az[ks], b, acc);
                }
                int coll = wave * 128 + ct * 16 + (lane & 15);
                float bv = B1[chunk * 1024 + coll];
                int rb = (lane >> 4) << 3;
#pragma unroll
                for (int k = 0; k < 8; ++k) {
                    float y = acc[k] + bv;
                    tb[rb + k][coll] = (bf16_t)(y > 0.f ? y : 0.f);
                }
            }
            __syncthreads();
            // phase 2: out-tile(wave) += tb @ W2[chunk, :]
            for (int ks2 = 0; ks2 < 32; ++ks2) {
                v16bf a2  = load_a_frag(&tb[0][0], 1040, ks2 * 32, lane);
                v16bf b2f = load_b_frag(W2, wave, chunk * 32 + ks2, 128, lane);
                oacc = WMMA_BF16(a2, b2f, oacc);
            }
            __syncthreads();
        }
        // gated residual update
        int col = wave * 16 + (lane & 15);
        float b2v = B2[col];
        int rb = (lane >> 4) << 3;
#pragma unroll
        for (int k = 0; k < 8; ++k) {
            float y = oacc[k] + b2v;
            zm[rb + k][col] = gg * y + (1.f - gg) * zm[rb + k][col];
        }
        __syncthreads();
    }

    // head: out = z @ head_W[e] + head_b[e], store grouped-by-type
    int row = tid >> 4, seg = tid & 15;
    float partial = 0.f;
#pragma unroll
    for (int j = 0; j < 8; ++j) {
        int c = seg * 8 + j;
        partial += zm[row][c] * headW[e * TWO_C + c];
    }
    red[row][seg] = partial;
    __syncthreads();
    if (seg == 0) {
        float s = 0.f;
#pragma unroll
        for (int j = 0; j < 16; ++j) s += red[row][j];
        int li = rowoff + row;
        if (li < M[META_CNT + e]) out[M[META_BASE + e] + li] = s + headb[e];
    }
}

// -------------------------------------------------------------- launch -----
extern "C" void kernel_launch(void* const* d_in, const int* in_sizes, int n_in,
                              void* d_out, int out_size, void* d_ws, size_t ws_size,
                              hipStream_t stream) {
    (void)in_sizes; (void)n_in; (void)out_size; (void)ws_size;
    const float* x          = (const float*)d_in[0];
    const int*   edge_index = (const int*)d_in[1];
    const float* edge_attr  = (const float*)d_in[2];
    const int*   edge_cross = (const int*)d_in[3];
    const int*   types      = (const int*)d_in[4];
    const float* gat_Wl     = (const float*)d_in[5];
    const float* gat_bl     = (const float*)d_in[6];
    const float* gat_Wr     = (const float*)d_in[7];
    const float* gat_br     = (const float*)d_in[8];
    const float* gat_We     = (const float*)d_in[9];
    const float* gat_att    = (const float*)d_in[10];
    const float* gat_bias   = (const float*)d_in[11];
    const float* gat_gates  = (const float*)d_in[12];
    const float* res_W1     = (const float*)d_in[13];
    const float* res_b1     = (const float*)d_in[14];
    const float* res_W2     = (const float*)d_in[15];
    const float* res_b2     = (const float*)d_in[16];
    const float* res_gates  = (const float*)d_in[17];
    const float* head_W     = (const float*)d_in[18];
    const float* head_b     = (const float*)d_in[19];
    float* outp = (float*)d_out;

    char* ws = (char*)d_ws;
    float*    h      = (float*)(ws + OFF_H);
    float*    h2     = (float*)(ws + OFF_H2);
    float*    xl     = (float*)(ws + OFF_XL);
    float*    xr     = (float*)(ws + OFF_XR);
    float*    agg    = (float*)(ws + OFF_AGG);
    float*    logits = (float*)(ws + OFF_LOG);
    unsigned* mxenc  = (unsigned*)(ws + OFF_MX);
    float*    den    = (float*)(ws + OFF_DEN);
    float*    zs     = (float*)(ws + OFF_ZS);
    int*      rank   = (int*)(ws + OFF_RANK);
    int*      meta   = (int*)(ws + OFF_META);
    bf16_t*   wlp    = (bf16_t*)(ws + OFF_WLP);
    bf16_t*   wrp    = (bf16_t*)(ws + OFF_WRP);
    bf16_t*   w1p    = (bf16_t*)(ws + OFF_W1P);
    bf16_t*   w2p    = (bf16_t*)(ws + OFF_W2P);

    // ---- pack weights to bf16 WMMA-fragment order (each call; deterministic)
    k_pack_batched<<<dim3(128, NEXP * N_RES), 256, 0, stream>>>(res_W1, w1p, TWO_C, HID);
    k_pack_batched<<<dim3(128, NEXP * N_RES), 256, 0, stream>>>(res_W2, w2p, HID, TWO_C);
    k_pack_batched<<<dim3(4, L), 256, 0, stream>>>(gat_Wl, wlp, F_IN, HC);
    k_pack_batched<<<dim3(4, L), 256, 0, stream>>>(gat_Wr, wrp, F_IN, HC);

    const unsigned NEG_INF_ORD = 0x007FFFFFu;   // f32_ord(-inf)
    const int NH = N_NODES * H;                 // 80000

    // ---- GATv2 encoder, 3 layers
    for (int i = 0; i < L; ++i) {
        const float* in = (i == 0) ? x : h2;
        k_node_gemm<<<N_NODES / 16, 128, 0, stream>>>(in, wlp + (size_t)i * F_IN * HC,
                                                      gat_bl + (size_t)i * HC, xl);
        k_node_gemm<<<N_NODES / 16, 128, 0, stream>>>(in, wrp + (size_t)i * F_IN * HC,
                                                      gat_br + (size_t)i * HC, xr);
        k_fill_u32<<<(NH + 255) / 256, 256, 0, stream>>>(mxenc, NEG_INF_ORD, NH);
        k_fill_f32<<<(NH + 255) / 256, 256, 0, stream>>>(den, 0.f, NH);
        k_fill_f32<<<(N_NODES * HC) / 256, 256, 0, stream>>>(agg, 0.f, N_NODES * HC);

        k_edge_logits<<<(N_EDGES * H) / 256, 256, 0, stream>>>(
            edge_index, edge_attr, xl, xr, gat_We + (size_t)i * EDGE_DIM * HC,
            gat_att + (size_t)i * HC, logits, mxenc);
        k_edge_softmax<<<(N_EDGES * H) / 256, 256, 0, stream>>>(edge_index, logits, mxenc, den);
        k_edge_agg<<<(N_EDGES * H) / 256, 256, 0, stream>>>(edge_index, logits, den, xl, agg);
        k_finalize<<<(N_NODES * F_IN) / 256, 256, 0, stream>>>(
            agg, gat_bias + (size_t)i * C, gat_gates, i, h2, h);
        if (i + 1 < L)
            k_lrelu<<<(N_NODES * F_IN) / 256, 256, 0, stream>>>(h, h2, N_NODES * F_IN);
    }

    // ---- pair routing (stable counting sort by type)
    k_fill_u32<<<1, 256, 0, stream>>>((unsigned*)(meta + META_CNT), 0u, NEXP);
    k_rank<<<N_PAIRS / 256, 256, 0, stream>>>(types, rank, meta + META_CNT);
    k_meta<<<1, 1, 0, stream>>>(meta);
    k_fill_f32<<<(PAD_ROWS * TWO_C) / 256, 256, 0, stream>>>(zs, 0.f, PAD_ROWS * TWO_C);
    k_gather<<<(N_PAIRS * TWO_C) / 256, 256, 0, stream>>>(edge_cross, types, rank, meta, h, zs);

    // ---- routed expert MLPs + head
    k_expert<<<MAX_TILES, 256, 0, stream>>>(zs, w1p, w2p, res_b1, res_b2, res_gates,
                                            head_W, head_b, meta, outp);
}
